// Transformer_51196010169063
// MI455X (gfx1250) — matrix operand order repaired and checked by reading, not compiled, not run on hardware
//
#include <hip/hip_runtime.h>
#include <cstdint>
#include <cstddef>

// ---------------- problem constants (from reference setup) ----------------
static constexpr int Bb  = 2;      // batch
static constexpr int Ss  = 1024;   // sequence length (enc == dec)
static constexpr int Dd  = 512;    // model dim
static constexpr int Ff  = 2048;   // ffn dim
static constexpr int Hh  = 8;      // heads
static constexpr int DHd = 64;     // head dim
static constexpr int Vv  = 16000;  // vocab
static constexpr int Pp  = Bb * Ss;

// ---------------- types ----------------
typedef __attribute__((ext_vector_type(16))) __bf16        v16bf;
typedef __attribute__((ext_vector_type(8)))  float         f32x8;
typedef __attribute__((ext_vector_type(4)))  unsigned int  u32x4;

union Frag16 { v16bf v; u32x4 q[2]; };

__device__ __forceinline__ unsigned short f2bf(float f) {
  unsigned int u = __float_as_uint(f);
  u += 0x7FFFu + ((u >> 16) & 1u);      // round-to-nearest-even
  return (unsigned short)(u >> 16);
}

// LDS byte offset of a __shared__ object (flat shared pointer: low 32 bits = LDS addr)
__device__ __forceinline__ unsigned lds_off(const void* p) {
  return (unsigned)(uintptr_t)p;
}

// CDNA5 async global->LDS copy (ASYNCcnt-tracked); per-lane 16B transfer.
__device__ __forceinline__ void async_load_b128(unsigned ldsaddr, const void* gaddr) {
  asm volatile("global_load_async_to_lds_b128 %0, %1, off"
               :: "v"(ldsaddr), "v"(gaddr)
               : "memory");
}
__device__ __forceinline__ void wait_async0() {
  asm volatile("s_wait_asynccnt 0" ::: "memory");
}

// ---------------- GEMM-NT, bf16 in / f32 out, WMMA 16x16x32 ----------------
// C[M,N] = alpha * A[M,K] * B[N,K]^T + bias[N]  (both operands K-contiguous rows)
// Batched over grid.z with (b,h) = (z/nh, z%nh) strides.
// Double-buffered LDS fed by async global->LDS loads; one barrier per k-step.
#define BM 128
#define BK 32

template <int BN_>
__global__ __launch_bounds__(256)
void gemm_bf16_nt(const unsigned short* __restrict__ A, int lda, long long sAb, long long sAh,
                  const unsigned short* __restrict__ Bm, int ldb, long long sBb, long long sBh,
                  float* __restrict__ C, int ldc, long long sCb, long long sCh,
                  int M, int N, int K, float alpha,
                  const float* __restrict__ bias, int relu, int nh)
{
  constexpr int JT  = BN_ / 32;               // 16-col subtiles per wave (wave covers BN_/2)
  constexpr int ACH = (BM  * BK) / (256 * 8); // A b128-chunks per thread (2)
  constexpr int BCH = (BN_ * BK) / (256 * 8); // B b128-chunks per thread (2 or 1)

  __shared__ alignas(16) unsigned short As[2][BM * BK];
  __shared__ alignas(16) unsigned short Bs[2][BN_ * BK];

  const int z  = blockIdx.z;
  const int zb = z / nh, zh = z % nh;
  A  += (size_t)zb * sAb + (size_t)zh * sAh;
  Bm += (size_t)zb * sBb + (size_t)zh * sBh;
  C  += (size_t)zb * sCb + (size_t)zh * sCh;

  const int tile_m = blockIdx.x * BM;
  const int tile_n = blockIdx.y * BN_;

  const int tid  = threadIdx.x;
  const int wave = tid >> 5, lane = tid & 31;
  const int wm = wave >> 1, wn = wave & 1;   // 4x2 waves: 32 rows x (BN_/2) cols each
  const int kh = lane >> 4, r16 = lane & 15; // lane-half / row-in-16

  f32x8 acc[2][JT];
#pragma unroll
  for (int i = 0; i < 2; ++i)
#pragma unroll
    for (int j = 0; j < JT; ++j)
      acc[i][j] = (f32x8){0.f, 0.f, 0.f, 0.f, 0.f, 0.f, 0.f, 0.f};

  // stage tile (k0) into LDS buffer p via async copies; chunk c -> row=c>>2, col=(c&3)*8
  auto stage = [&](int p, int k0) {
#pragma unroll
    for (int t = 0; t < ACH; ++t) {
      int c = tid + t * 256, row = c >> 2, col = (c & 3) * 8;
      async_load_b128(lds_off(&As[p][row * BK + col]),
                      A + (size_t)(tile_m + row) * lda + k0 + col);
    }
#pragma unroll
    for (int t = 0; t < BCH; ++t) {
      int c = tid + t * 256, row = c >> 2, col = (c & 3) * 8;
      async_load_b128(lds_off(&Bs[p][row * BK + col]),
                      Bm + (size_t)(tile_n + row) * ldb + k0 + col);
    }
  };

  stage(0, 0);
  wait_async0();
  __syncthreads();

  int p = 0;
  for (int k0 = 0; k0 < K; k0 += BK) {
    const bool has_next = (k0 + BK < K);
    if (has_next) stage(p ^ 1, k0 + BK);   // latency hides behind WMMA below

    // A frag (16x32 bf16): lane-half kh -> K = {kh*8..+8} and {16+kh*8..+8}
    Frag16 af[2];
#pragma unroll
    for (int i = 0; i < 2; ++i) {
      int row = wm * 32 + i * 16 + r16;
      af[i].q[0] = *(const u32x4*)&As[p][row * BK + kh * 8];
      af[i].q[1] = *(const u32x4*)&As[p][row * BK + 16 + kh * 8];
    }
    // B frag (32x16 bf16): lane-half kh -> K = {kh*8..+8} and {16+kh*8..+8}; reuse across both A subtiles
#pragma unroll
    for (int j = 0; j < JT; ++j) {
      int col = wn * (BN_ / 2) + j * 16 + r16;
      Frag16 bfr;
      bfr.q[0] = *(const u32x4*)&Bs[p][col * BK + kh * 8];
      bfr.q[1] = *(const u32x4*)&Bs[p][col * BK + 16 + kh * 8];
#pragma unroll
      for (int i = 0; i < 2; ++i)
        acc[i][j] = __builtin_amdgcn_wmma_f32_16x16x32_bf16(
            false, af[i].v, false, bfr.v, (short)0, acc[i][j], false, false);
    }

    if (has_next) {
      wait_async0();       // own async writes to LDS[p^1] landed
      __syncthreads();     // everyone's landed; also fences reads of LDS[p]
      p ^= 1;
    }
  }

  // C/D layout: lane half kh -> rows {v + kh*8}, col = r16
#pragma unroll
  for (int i = 0; i < 2; ++i) {
#pragma unroll
    for (int j = 0; j < JT; ++j) {
      int col = tile_n + wn * (BN_ / 2) + j * 16 + r16;
      float bv = bias ? bias[col] : 0.0f;
#pragma unroll
      for (int v = 0; v < 8; ++v) {
        int row = tile_m + wm * 32 + i * 16 + kh * 8 + v;
        float val = acc[i][j][v] * alpha + bv;
        if (relu) val = fmaxf(val, 0.0f);
        C[(size_t)row * ldc + col] = val;
      }
    }
  }
}

// ---------------- embedding + sinusoidal PE ----------------
__global__ __launch_bounds__(256)
void embed_pe(const int* __restrict__ ids, const float* __restrict__ embw,
              float* __restrict__ out)
{
  int p = blockIdx.x;                 // 0..Pp-1  (b*Ss + s)
  int s = p % Ss;
  int id = ids[p];
  const float* e = embw + (size_t)id * Dd;
  float* o = out + (size_t)p * Dd;
  for (int d = threadIdx.x; d < Dd; d += 256) {
    int i = d >> 1;
    float div = __powf(10000.0f, (2.0f * (float)i) / (float)Dd);
    float ang = (float)s / div;
    float pe  = (d & 1) ? __cosf(ang) : __sinf(ang);
    o[d] = e[d] + pe;
  }
}

// ---------------- converts ----------------
__global__ __launch_bounds__(256)
void cvt_f32_bf16(const float* __restrict__ s, unsigned short* __restrict__ d, size_t n)
{
  size_t i = (size_t)blockIdx.x * blockDim.x + threadIdx.x;
  if (i < n) d[i] = f2bf(s[i]);
}

// W [K,N] f32 (row-major) -> Wt [N,K] bf16 (K-contiguous rows)
__global__ __launch_bounds__(256)
void cvtT_f32_bf16(const float* __restrict__ Wm, unsigned short* __restrict__ Wtd,
                   int Kd, int Nd)
{
  size_t i = (size_t)blockIdx.x * blockDim.x + threadIdx.x;
  if (i < (size_t)Kd * Nd) {
    int k = (int)(i % Kd);
    int n = (int)(i / Kd);
    Wtd[i] = f2bf(Wm[(size_t)k * Nd + n]);
  }
}

// V f32 [P, D] -> Vt bf16 [B, H, dh, S]   (per-head transpose for att*V GEMM)
__global__ __launch_bounds__(256)
void transpose_heads(const float* __restrict__ Vf, unsigned short* __restrict__ Vt)
{
  size_t i = (size_t)blockIdx.x * 256 + threadIdx.x;  // over Bb*Hh*DHd*Ss
  int n = (int)(i % Ss);
  size_t t = i / Ss;
  int kd = (int)(t % DHd); t /= DHd;
  int h  = (int)(t % Hh);
  int b  = (int)(t / Hh);
  Vt[i] = f2bf(Vf[((size_t)(b * Ss + n)) * Dd + h * DHd + kd]);
}

// ---------------- masked softmax, f32 scores -> bf16 att ----------------
__global__ __launch_bounds__(256)
void softmax_mask_bf16(const float* __restrict__ scores, unsigned short* __restrict__ att,
                       const int* __restrict__ mask, int causal)
{
  int rowid = blockIdx.x;             // over Bb*Hh*Ss
  int m = rowid % Ss;
  int b = rowid / (Hh * Ss);
  const float* srow = scores + (size_t)rowid * Ss;
  unsigned short* arow = att + (size_t)rowid * Ss;
  int tid = threadIdx.x;

  float v[4];
  float mx = -3.4e38f;
#pragma unroll
  for (int u = 0; u < 4; ++u) {
    int n = tid + u * 256;
    bool ok = (!causal || n <= m) && (mask[b * Ss + n] != 0);
    float s = ok ? srow[n] : -3.4e38f;
    v[u] = s;
    mx = fmaxf(mx, s);
  }
  __shared__ float red[8];
  for (int off = 16; off > 0; off >>= 1) mx = fmaxf(mx, __shfl_down(mx, off));
  if ((tid & 31) == 0) red[tid >> 5] = mx;
  __syncthreads();
  float MX = red[0];
#pragma unroll
  for (int i = 1; i < 8; ++i) MX = fmaxf(MX, red[i]);
  __syncthreads();

  float sum = 0.f;
#pragma unroll
  for (int u = 0; u < 4; ++u) {
    float e = (v[u] > -1e37f) ? __expf(v[u] - MX) : 0.f;
    v[u] = e;
    sum += e;
  }
  for (int off = 16; off > 0; off >>= 1) sum += __shfl_down(sum, off);
  if ((tid & 31) == 0) red[tid >> 5] = sum;
  __syncthreads();
  float SUM = 0.f;
#pragma unroll
  for (int i = 0; i < 8; ++i) SUM += red[i];
  float inv = 1.0f / SUM;
#pragma unroll
  for (int u = 0; u < 4; ++u) {
    int n = tid + u * 256;
    arow[n] = f2bf(v[u] * inv);
  }
}

// ---------------- out = LayerNorm(a + b) * g + beta  (sigma = std + eps) -----
__global__ __launch_bounds__(256)
void add_layernorm(const float* __restrict__ a, const float* __restrict__ b,
                   const float* __restrict__ g, const float* __restrict__ be,
                   float* __restrict__ out)
{
  int row = blockIdx.x, tid = threadIdx.x;
  const float* ar = a + (size_t)row * Dd;
  const float* br = b + (size_t)row * Dd;
  float x0 = ar[tid] + br[tid];
  float x1 = ar[tid + 256] + br[tid + 256];

  __shared__ float red[8];
  float s = x0 + x1;
  for (int off = 16; off > 0; off >>= 1) s += __shfl_down(s, off);
  if ((tid & 31) == 0) red[tid >> 5] = s;
  __syncthreads();
  float tot = 0.f;
#pragma unroll
  for (int i = 0; i < 8; ++i) tot += red[i];
  float mu = tot / (float)Dd;
  __syncthreads();

  float d0 = x0 - mu, d1 = x1 - mu;
  float s2 = d0 * d0 + d1 * d1;
  for (int off = 16; off > 0; off >>= 1) s2 += __shfl_down(s2, off);
  if ((tid & 31) == 0) red[tid >> 5] = s2;
  __syncthreads();
  float tot2 = 0.f;
#pragma unroll
  for (int i = 0; i < 8; ++i) tot2 += red[i];
  float sigma = sqrtf(tot2 / (float)Dd) + 1e-6f;

  float* orow = out + (size_t)row * Dd;
  orow[tid]       = d0 / sigma * g[tid]       + be[tid];
  orow[tid + 256] = d1 / sigma * g[tid + 256] + be[tid + 256];
}

// ---------------- host orchestration ----------------
extern "C" void kernel_launch(void* const* d_in, const int* in_sizes, int n_in,
                              void* d_out, int out_size, void* d_ws, size_t ws_size,
                              hipStream_t stream)
{
  (void)in_sizes; (void)n_in; (void)out_size; (void)ws_size;

  const float* emb  = (const float*)d_in[0];
  const float* eWq  = (const float*)d_in[1];
  const float* eWk  = (const float*)d_in[2];
  const float* eWv  = (const float*)d_in[3];
  const float* eWo  = (const float*)d_in[4];
  const float* eW1  = (const float*)d_in[5];
  const float* eb1  = (const float*)d_in[6];
  const float* eW2  = (const float*)d_in[7];
  const float* eb2  = (const float*)d_in[8];
  const float* eln1g = (const float*)d_in[9];
  const float* eln1b = (const float*)d_in[10];
  const float* eln2g = (const float*)d_in[11];
  const float* eln2b = (const float*)d_in[12];
  const float* dWq1 = (const float*)d_in[13];
  const float* dWk1 = (const float*)d_in[14];
  const float* dWv1 = (const float*)d_in[15];
  const float* dWo1 = (const float*)d_in[16];
  const float* dWq2 = (const float*)d_in[17];
  const float* dWk2 = (const float*)d_in[18];
  const float* dWv2 = (const float*)d_in[19];
  const float* dWo2 = (const float*)d_in[20];
  const float* dW1  = (const float*)d_in[21];
  const float* db1  = (const float*)d_in[22];
  const float* dW2  = (const float*)d_in[23];
  const float* db2  = (const float*)d_in[24];
  const float* dln1g = (const float*)d_in[25];
  const float* dln1b = (const float*)d_in[26];
  const float* dln2g = (const float*)d_in[27];
  const float* dln2b = (const float*)d_in[28];
  const float* dln3g = (const float*)d_in[29];
  const float* dln3b = (const float*)d_in[30];
  const int* eids  = (const int*)d_in[31];
  const int* dids  = (const int*)d_in[32];
  const int* emask = (const int*)d_in[33];
  const int* dmask = (const int*)d_in[34];

  // ---- workspace carve (~162 MB) ----
  char* wsp = (char*)d_ws;
  auto take = [&](size_t bytes) -> char* {
    char* p = wsp; wsp += (bytes + 255) & ~(size_t)255; return p;
  };
  float* x    = (float*)take((size_t)Pp * Dd * 4);
  float* y    = (float*)take((size_t)Pp * Dd * 4);
  float* tmp1 = (float*)take((size_t)Pp * Ff * 4);
  float* tmp2 = (float*)take((size_t)Pp * Dd * 4);
  float* tmp3 = (float*)take((size_t)Pp * Dd * 4);
  unsigned short* Abf  = (unsigned short*)take((size_t)Pp * Ff * 2);
  unsigned short* Wt   = (unsigned short*)take((size_t)Dd * Ff * 2);
  unsigned short* Qbf  = (unsigned short*)take((size_t)Pp * Dd * 2);
  unsigned short* Kbf  = (unsigned short*)take((size_t)Pp * Dd * 2);
  unsigned short* Vt   = (unsigned short*)take((size_t)Bb * Hh * DHd * Ss * 2);
  unsigned short* encb = (unsigned short*)take((size_t)Pp * Dd * 2);
  unsigned short* embb = (unsigned short*)take((size_t)Vv * Dd * 2);
  float* scores        = (float*)take((size_t)Bb * Hh * Ss * Ss * 4);
  unsigned short* att  = (unsigned short*)take((size_t)Bb * Hh * Ss * Ss * 2);

  // ---- launch helpers ----
  auto cvt = [&](const float* s, unsigned short* d, size_t n) {
    cvt_f32_bf16<<<dim3((unsigned)((n + 255) / 256)), 256, 0, stream>>>(s, d, n);
  };
  auto cvtT = [&](const float* Wm, unsigned short* Wtd, int Kd, int Nd) {
    size_t n = (size_t)Kd * Nd;
    cvtT_f32_bf16<<<dim3((unsigned)((n + 255) / 256)), 256, 0, stream>>>(Wm, Wtd, Kd, Nd);
  };
  auto gemm = [&](const unsigned short* Aa, int lda, long long sAb, long long sAh,
                  const unsigned short* Ba, int ldb, long long sBb, long long sBh,
                  float* Ca, int ldc, long long sCb, long long sCh,
                  int M, int N, int K, float alpha, const float* bias, int relu,
                  int batch, int nh) {
    if (N % 128 == 0) {
      dim3 g(M / BM, N / 128, batch);
      gemm_bf16_nt<128><<<g, 256, 0, stream>>>(Aa, lda, sAb, sAh, Ba, ldb, sBb, sBh,
                                               Ca, ldc, sCb, sCh, M, N, K, alpha, bias, relu, nh);
    } else {
      dim3 g(M / BM, N / 64, batch);
      gemm_bf16_nt<64><<<g, 256, 0, stream>>>(Aa, lda, sAb, sAh, Ba, ldb, sBb, sBh,
                                              Ca, ldc, sCb, sCh, M, N, K, alpha, bias, relu, nh);
    }
  };
  auto lnadd = [&](const float* a, const float* b, const float* g, const float* be, float* o) {
    add_layernorm<<<Pp, 256, 0, stream>>>(a, b, g, be, o);
  };

  // Full MHA: out = proj(softmax(mask(Q Kt / sqrt(dh))) V) ; out -> outp (f32 [P,D])
  auto mha = [&](const unsigned short* qin, const unsigned short* kvin,
                 const float* Wq, const float* Wk, const float* Wv, const float* Wo,
                 int causal, const int* mask, float* outp) {
    cvtT(Wq, Wt, Dd, Dd);
    gemm(qin, Dd, 0, 0, Wt, Dd, 0, 0, tmp3, Dd, 0, 0, Pp, Dd, Dd, 1.f, nullptr, 0, 1, 1);
    cvt(tmp3, Qbf, (size_t)Pp * Dd);
    cvtT(Wk, Wt, Dd, Dd);
    gemm(kvin, Dd, 0, 0, Wt, Dd, 0, 0, tmp3, Dd, 0, 0, Pp, Dd, Dd, 1.f, nullptr, 0, 1, 1);
    cvt(tmp3, Kbf, (size_t)Pp * Dd);
    cvtT(Wv, Wt, Dd, Dd);
    gemm(kvin, Dd, 0, 0, Wt, Dd, 0, 0, tmp3, Dd, 0, 0, Pp, Dd, Dd, 1.f, nullptr, 0, 1, 1);
    transpose_heads<<<(Bb * Hh * DHd * Ss) / 256, 256, 0, stream>>>(tmp3, Vt);
    // scores[b,h,m,n] = Q . K / 8
    gemm(Qbf, Dd, (long long)Ss * Dd, DHd,
         Kbf, Dd, (long long)Ss * Dd, DHd,
         scores, Ss, (long long)Hh * Ss * Ss, (long long)Ss * Ss,
         Ss, Ss, DHd, 0.125f, nullptr, 0, Bb * Hh, Hh);
    softmax_mask_bf16<<<Bb * Hh * Ss, 256, 0, stream>>>(scores, att, mask, causal);
    // out[b,m,h*64+kd] = att . V   (N=64 -> BN=64 kernel variant)
    gemm(att, Ss, (long long)Hh * Ss * Ss, (long long)Ss * Ss,
         Vt, Ss, (long long)Hh * DHd * Ss, (long long)DHd * Ss,
         tmp3, Dd, (long long)Ss * Dd, DHd,
         Ss, DHd, Ss, 1.f, nullptr, 0, Bb * Hh, Hh);
    cvt(tmp3, Abf, (size_t)Pp * Dd);   // qin(=Abf) no longer needed by now
    cvtT(Wo, Wt, Dd, Dd);
    gemm(Abf, Dd, 0, 0, Wt, Dd, 0, 0, outp, Dd, 0, 0, Pp, Dd, Dd, 1.f, nullptr, 0, 1, 1);
  };

  auto ffn = [&](float* xin, const float* W1, const float* b1p,
                 const float* W2, const float* b2p) {
    cvt(xin, Abf, (size_t)Pp * Dd);
    cvtT(W1, Wt, Dd, Ff);
    gemm(Abf, Dd, 0, 0, Wt, Dd, 0, 0, tmp1, Ff, 0, 0, Pp, Ff, Dd, 1.f, b1p, 1, 1, 1);
    cvt(tmp1, Abf, (size_t)Pp * Ff);
    cvtT(W2, Wt, Ff, Dd);
    gemm(Abf, Ff, 0, 0, Wt, Ff, 0, 0, tmp2, Dd, 0, 0, Pp, Dd, Ff, 1.f, b2p, 0, 1, 1);
  };

  // ---- tied-head embedding in bf16 (rows are already [vocab, K]) ----
  cvt(emb, embb, (size_t)Vv * Dd);

  // ================= encoder =================
  embed_pe<<<Pp, 256, 0, stream>>>(eids, emb, x);
  for (int l = 0; l < 2; ++l) {
    cvt(x, Abf, (size_t)Pp * Dd);
    mha(Abf, Abf,
        eWq + (size_t)l * Dd * Dd, eWk + (size_t)l * Dd * Dd,
        eWv + (size_t)l * Dd * Dd, eWo + (size_t)l * Dd * Dd,
        0, emask, tmp2);
    lnadd(tmp2, x, eln1g + l * Dd, eln1b + l * Dd, x);
    ffn(x, eW1 + (size_t)l * Dd * Ff, eb1 + l * Ff,
           eW2 + (size_t)l * Ff * Dd, eb2 + l * Dd);
    lnadd(x, tmp2, eln2g + l * Dd, eln2b + l * Dd, x);
  }
  cvt(x, encb, (size_t)Pp * Dd);   // enc_out (bf16) for cross-attention K/V

  // ================= decoder =================
  embed_pe<<<Pp, 256, 0, stream>>>(dids, emb, y);
  for (int l = 0; l < 2; ++l) {
    cvt(y, Abf, (size_t)Pp * Dd);
    mha(Abf, Abf,
        dWq1 + (size_t)l * Dd * Dd, dWk1 + (size_t)l * Dd * Dd,
        dWv1 + (size_t)l * Dd * Dd, dWo1 + (size_t)l * Dd * Dd,
        1, dmask, tmp2);
    lnadd(tmp2, y, dln1g + l * Dd, dln1b + l * Dd, y);
    cvt(y, Abf, (size_t)Pp * Dd);
    mha(Abf, encb,
        dWq2 + (size_t)l * Dd * Dd, dWk2 + (size_t)l * Dd * Dd,
        dWv2 + (size_t)l * Dd * Dd, dWo2 + (size_t)l * Dd * Dd,
        0, emask, tmp2);
    lnadd(y, tmp2, dln2g + l * Dd, dln2b + l * Dd, y);
    ffn(y, dW1 + (size_t)l * Dd * Ff, db1 + l * Ff,
           dW2 + (size_t)l * Ff * Dd, db2 + l * Dd);
    lnadd(y, tmp2, dln3g + l * Dd, dln3b + l * Dd, y);
  }

  // ================= tied logits head =================
  cvt(y, Abf, (size_t)Pp * Dd);
  gemm(Abf, Dd, 0, 0, embb, Dd, 0, 0, (float*)d_out, Vv, 0, 0,
       Pp, Vv, Dd, 1.f, nullptr, 0, 1, 1);
}
